// ProtoNet_2087354106027
// MI455X (gfx1250) — matrix operand order repaired and checked by reading, not compiled, not run on hardware
//
#include <hip/hip_runtime.h>

// ---------------------------------------------------------------------------
// ProtoNet episode: n_way=512, n_support=16, n_query=64, d=4096
//   prototypes = mean(support)            (512 x 4096)
//   query      = (32768 x 4096)
//   dist[q,w]  = sqrt(max(q2 + p2 - 2*qp, 0))
//   log_p = log_softmax(-dist, axis=1); y_hat = argmax; loss = -mean(log_p)
// Heavy op: 32768x512x4096 GEMM -> bf16 WMMA (v_wmma_f32_16x16x32_bf16),
// fused distance + log-softmax + argmax + loss epilogue (distances stay in LDS).
// ---------------------------------------------------------------------------

#define NWAY     512
#define NSUP     16
#define NQRY     64
#define DDIM     4096
#define TOTQ     (NWAY * NQRY)   // 32768
#define MT       32              // query rows per block
#define KC       32              // bf16 WMMA K per step

typedef __attribute__((ext_vector_type(16))) __bf16 v16bf;
typedef __attribute__((ext_vector_type(8)))  float  v8f;

union Frag {
    uint4  u[2];   // two 16B global loads
    v16bf  v;      // WMMA operand view
};

__device__ __forceinline__ unsigned int f2bf_bits(float f) {
    unsigned int u = __float_as_uint(f);
    u += 0x7FFFu + ((u >> 16) & 1u);           // round-to-nearest-even
    return u >> 16;
}
__device__ __forceinline__ float bf2f_bits(unsigned int h) {
    return __uint_as_float(h << 16);
}
// pack two fp32 -> two bf16 in one dword (lo = a, hi = b); return rounded floats
__device__ __forceinline__ unsigned int pack2(float a, float b, float* ra, float* rb) {
    unsigned int ba = f2bf_bits(a), bb = f2bf_bits(b);
    *ra = bf2f_bits(ba);
    *rb = bf2f_bits(bb);
    return ba | (bb << 16);
}

// ---------------------------------------------------------------------------
// Kernel 1: prototypes = mean over 16 support rows; write bf16 proto + |p|^2.
// One block per class, float4 loads / uint2 packed bf16 stores.
// images: (512, 80, 4096) fp32.
// ---------------------------------------------------------------------------
__global__ __launch_bounds__(256) void proto_prep_kernel(
    const float* __restrict__ images,
    unsigned short* __restrict__ pbf,
    float* __restrict__ p2)
{
    const int w   = blockIdx.x;
    const int tid = threadIdx.x;
    const float* base = images + (size_t)w * (NSUP + NQRY) * DDIM;

    __shared__ float red[256];
    float pp = 0.f;
    #pragma unroll
    for (int it = 0; it < DDIM / (256 * 4); ++it) {     // 4 iterations
        const int d = (tid + it * 256) * 4;
        float4 s = make_float4(0.f, 0.f, 0.f, 0.f);
        #pragma unroll
        for (int j = 0; j < NSUP; ++j) {
            float4 v = *(const float4*)(base + (size_t)j * DDIM + d);
            s.x += v.x; s.y += v.y; s.z += v.z; s.w += v.w;
        }
        const float inv = 1.f / (float)NSUP;
        s.x *= inv; s.y *= inv; s.z *= inv; s.w *= inv;
        float rx, ry, rz, rw;
        uint2 packed;
        packed.x = pack2(s.x, s.y, &rx, &ry);
        packed.y = pack2(s.z, s.w, &rz, &rw);
        *(uint2*)(pbf + (size_t)w * DDIM + d) = packed;
        pp += rx * rx + ry * ry + rz * rz + rw * rw;    // p2 from bf16-rounded
    }
    red[tid] = pp;
    __syncthreads();
    for (int s = 128; s > 0; s >>= 1) {
        if (tid < s) red[tid] += red[tid + s];
        __syncthreads();
    }
    if (tid == 0) p2[w] = red[0];
}

// ---------------------------------------------------------------------------
// Kernel 2: bf16 copy of query rows + |q|^2. One block per query row.
// float4 loads / uint2 packed bf16 stores.
// ---------------------------------------------------------------------------
__global__ __launch_bounds__(256) void query_prep_kernel(
    const float* __restrict__ images,
    unsigned short* __restrict__ qbf,
    float* __restrict__ q2)
{
    const int q   = blockIdx.x;
    const int tid = threadIdx.x;
    const int w   = q >> 6;            // / NQRY
    const int j   = q & 63;            // % NQRY
    const float* src = images + ((size_t)w * (NSUP + NQRY) + NSUP + j) * DDIM;

    __shared__ float red[256];
    float qq = 0.f;
    #pragma unroll
    for (int it = 0; it < DDIM / (256 * 4); ++it) {     // 4 iterations
        const int d = (tid + it * 256) * 4;
        float4 v = *(const float4*)(src + d);
        float rx, ry, rz, rw;
        uint2 packed;
        packed.x = pack2(v.x, v.y, &rx, &ry);
        packed.y = pack2(v.z, v.w, &rz, &rw);
        *(uint2*)(qbf + (size_t)q * DDIM + d) = packed;
        qq += rx * rx + ry * ry + rz * rz + rw * rw;    // q2 from bf16-rounded
    }
    red[tid] = qq;
    __syncthreads();
    for (int s = 128; s > 0; s >>= 1) {
        if (tid < s) red[tid] += red[tid + s];
        __syncthreads();
    }
    if (tid == 0) q2[q] = red[0];
}

// ---------------------------------------------------------------------------
// Kernel 3: fused bf16 WMMA GEMM (qp) + distance + log-softmax + argmax + loss.
// Block = 256 threads = 8 waves. Block tile: 32 query rows x all 512 classes.
// Wave tile: 2 M-tiles x 4 N-tiles of 16x16 (wave's N slice = 64 columns).
// K streamed in steps of 32 (v_wmma_f32_16x16x32_bf16), 8 WMMA per step.
// ---------------------------------------------------------------------------
__global__ __launch_bounds__(256) void gemm_softmax_kernel(
    const unsigned short* __restrict__ qbf,   // TOTQ x DDIM bf16
    const unsigned short* __restrict__ pbf,   // NWAY x DDIM bf16
    const float* __restrict__ q2,
    const float* __restrict__ p2,
    float* __restrict__ out)                  // [0]=loss, [1..TOTQ]=y_hat
{
    __shared__ float xs[MT][NWAY + 4];        // -dist (padded: 66 KB)
    __shared__ float q2s[MT];
    __shared__ float p2s[NWAY];

    const int tid  = threadIdx.x;
    const int wave = tid >> 5;
    const int lane = tid & 31;
    const int hl   = lane >> 4;               // lane half (0/1)
    const int l16  = lane & 15;
    const int q0   = blockIdx.x * MT;
    const int n0   = wave * 64;               // this wave's N slice

    if (tid < MT) q2s[tid] = q2[q0 + tid];
    for (int i = tid; i < NWAY; i += 256) p2s[i] = p2[i];

    const unsigned short* aBase = qbf + (size_t)q0 * DDIM;

    v8f acc[2][4] = {};

    for (int k = 0; k < DDIM; k += KC) {
        Frag a[2], b[4];
        // A (query) 16x32 bf16 fragments: lane half selects K interleave.
        #pragma unroll
        for (int mt = 0; mt < 2; ++mt) {
            const unsigned short* ap =
                aBase + (size_t)(mt * 16 + l16) * DDIM + k + hl * 8;
            a[mt].u[0] = *(const uint4*)(ap);        // K = kb .. kb+7
            a[mt].u[1] = *(const uint4*)(ap + 16);   // K = kb+16 .. kb+23
            if (k + KC < DDIM)                       // streamed once: near prefetch
                __builtin_prefetch((const void*)(ap + KC), 0, 3);
        }
        // B (proto) 32x16 bf16 fragments: lane = N, half selects K interleave (same as A).
        #pragma unroll
        for (int nt = 0; nt < 4; ++nt) {
            const unsigned short* bp =
                pbf + (size_t)(n0 + nt * 16 + l16) * DDIM + k + hl * 8;
            b[nt].u[0] = *(const uint4*)(bp);
            b[nt].u[1] = *(const uint4*)(bp + 16);
            if (k + KC < DDIM)                       // L2-resident: far prefetch
                __builtin_prefetch((const void*)(bp + KC), 0, 2);
        }
        #pragma unroll
        for (int mt = 0; mt < 2; ++mt)
            #pragma unroll
            for (int nt = 0; nt < 4; ++nt)
                acc[mt][nt] = __builtin_amdgcn_wmma_f32_16x16x32_bf16(
                    false, a[mt].v, false, b[nt].v,
                    (short)0, acc[mt][nt], false, false);
    }

    // Epilogue 1: qp -> -dist into LDS.
    // C layout: VGPR r: lanes 0-15 -> M=r, lanes 16-31 -> M=8+r; N = lane&15.
    #pragma unroll
    for (int mt = 0; mt < 2; ++mt) {
        #pragma unroll
        for (int nt = 0; nt < 4; ++nt) {
            #pragma unroll
            for (int r = 0; r < 8; ++r) {
                const int m = mt * 16 + r + hl * 8;
                const int n = n0 + nt * 16 + l16;
                float d2 = q2s[m] + p2s[n] - 2.f * acc[mt][nt][r];
                xs[m][n] = -__builtin_sqrtf(fmaxf(d2, 0.f));
            }
        }
    }
    __syncthreads();

    // Epilogue 2: per-row log-softmax / argmax / loss. 4 rows per wave.
    float lossAcc = 0.f;
    #pragma unroll
    for (int i = 0; i < 4; ++i) {
        const int m = wave * 4 + i;
        float vals[16];
        float mx = -3.402823466e38f;
        int   ai = 0;
        #pragma unroll
        for (int jj = 0; jj < 16; ++jj) {
            float v = xs[m][lane + 32 * jj];
            vals[jj] = v;
            if (v > mx) { mx = v; ai = lane + 32 * jj; }
        }
        #pragma unroll
        for (int off = 16; off >= 1; off >>= 1) {
            float omx = __shfl_xor(mx, off);
            int   oai = __shfl_xor(ai, off);
            if (omx > mx || (omx == mx && oai < ai)) { mx = omx; ai = oai; }
        }
        float se = 0.f, sx = 0.f;
        #pragma unroll
        for (int jj = 0; jj < 16; ++jj) {
            se += __expf(vals[jj] - mx);
            sx += vals[jj];
        }
        #pragma unroll
        for (int off = 16; off >= 1; off >>= 1) {
            se += __shfl_xor(se, off);
            sx += __shfl_xor(sx, off);
        }
        float lse = mx + __logf(se);
        // row contribution to -mean(log_p): sum_n (lse - x) = NWAY*lse - sx
        lossAcc += (float)NWAY * lse - sx;
        if (lane == 0) out[1 + q0 + m] = (float)ai;   // y_hat
    }
    if (lane == 0)
        atomicAdd(&out[0], lossAcc * (1.f / ((float)TOTQ * (float)NWAY)));
}

// ---------------------------------------------------------------------------
extern "C" void kernel_launch(void* const* d_in, const int* in_sizes, int n_in,
                              void* d_out, int out_size, void* d_ws, size_t ws_size,
                              hipStream_t stream) {
    (void)in_sizes; (void)n_in; (void)out_size; (void)ws_size;
    const float* images = (const float*)d_in[0];   // (512, 80, 4096) fp32
    float* out = (float*)d_out;                    // [0]=loss, [1..32768]=y_hat

    // Workspace layout (~260.2 MB):
    unsigned short* qbf = (unsigned short*)d_ws;                 // 256 MB
    unsigned short* pbf = qbf + (size_t)TOTQ * DDIM;             // 4 MB
    float* q2 = (float*)(pbf + (size_t)NWAY * DDIM);             // 128 KB
    float* p2 = q2 + TOTQ;                                       // 2 KB

    hipMemsetAsync(d_out, 0, sizeof(float), stream);             // zero loss acc

    proto_prep_kernel<<<NWAY, 256, 0, stream>>>(images, pbf, p2);
    query_prep_kernel<<<TOTQ, 256, 0, stream>>>(images, qbf, q2);
    gemm_softmax_kernel<<<TOTQ / MT, 256, 0, stream>>>(qbf, pbf, q2, p2, out);
}
